// CrossAttentionHead_3547642987206
// MI455X (gfx1250) — matrix operand order repaired and checked by reading, not compiled, not run on hardware
//
#include <hip/hip_runtime.h>

// ---------------- CDNA5 WMMA types ----------------
typedef __attribute__((ext_vector_type(16))) __bf16 v16bf;
typedef __attribute__((ext_vector_type(8)))  float  v8f;
typedef __attribute__((ext_vector_type(4)))  unsigned int v4u;

#define WMMA_BF16(A, Bm, C) \
  __builtin_amdgcn_wmma_f32_16x16x32_bf16(false, (A), false, (Bm), (short)0, (C), false, false)

// Problem dimensions (from reference setup_inputs)
#define BATCH 4
#define LQ    4096
#define LKV   4096
#define EDIM  1024
#define HDIM  128
#define PROW  72   // padded LDS row stride (shorts): 144B avoids b128 bank conflicts

// ---------------- helpers ----------------
static __device__ __forceinline__ unsigned short f2bfu(float f) {
  unsigned u = __builtin_bit_cast(unsigned, f);
  u += 0x7FFFu + ((u >> 16) & 1u);            // RNE
  return (unsigned short)(u >> 16);
}
// pack two f32 -> dword of 2 bf16 (RNE round, then v_perm_b32 byte-select)
static __device__ __forceinline__ unsigned pk2(float lo, float hi) {
  unsigned a = __builtin_bit_cast(unsigned, lo);
  unsigned b = __builtin_bit_cast(unsigned, hi);
  a += 0x7FFFu + ((a >> 16) & 1u);
  b += 0x7FFFu + ((b >> 16) & 1u);
  return __builtin_amdgcn_perm(b, a, 0x07060302); // {b[31:16], a[31:16]}
}
// fragment = two 16B chunks, bit-cast directly (no element shuffling)
static __device__ __forceinline__ v16bf ldfrag(const unsigned short* p0,
                                               const unsigned short* p1) {
  union { v16bf v; v4u u[2]; } t;
  t.u[0] = *(const v4u*)p0;
  t.u[1] = *(const v4u*)p1;
  return t.v;
}

// ---------------- Stage 0: f32 -> bf16 streaming convert (RNE, optional scale) ----
__global__ __launch_bounds__(256) void xattn_cvt_bf16(
    const float* __restrict__ src, unsigned short* __restrict__ dst,
    float mul)
{
  size_t i = ((size_t)blockIdx.x * 256 + threadIdx.x) * 8;
  float4 a = *(const float4*)(src + i);
  float4 b = *(const float4*)(src + i + 4);
  v4u o;
  o.x = pk2(a.x * mul, a.y * mul);
  o.y = pk2(a.z * mul, a.w * mul);
  o.z = pk2(b.x * mul, b.y * mul);
  o.w = pk2(b.z * mul, b.w * mul);
  *(v4u*)(dst + i) = o;
}

// ---------------- Stage 1: QKV projection (bf16 in, bf16 out) ----------------
// grid.x = (B*L)/32/4, grid.y = 3 (q,k,v). Each wave: 32 rows x 128 cols,
// K = EDIM reduced in steps of 32. 16 WMMAs per K-step, 20 b128 loads.
__global__ __launch_bounds__(128) void xattn_qkv_proj(
    const unsigned short* __restrict__ xb, const unsigned short* __restrict__ zb,
    const unsigned short* __restrict__ Wb,   // [3][HDIM][EDIM] (Wq pre-scaled by sqrt(H))
    unsigned short* __restrict__ qb, unsigned short* __restrict__ kb,
    unsigned short* __restrict__ vT)
{
  const int lane = threadIdx.x & 31;
  const int wave = threadIdx.x >> 5;
  const int m0   = (blockIdx.x * 4 + wave) * 32;   // 32-row tile
  const int which = blockIdx.y;                    // 0:q 1:k 2:v
  const unsigned short* src = which ? zb : xb;
  const unsigned short* W   = Wb + (size_t)which * HDIM * EDIM;

  const int sel = lane >> 4;
  const int col = lane & 15;

  v8f acc[2][8];
#pragma unroll
  for (int h = 0; h < 2; ++h)
#pragma unroll
    for (int n = 0; n < 8; ++n) acc[h][n] = (v8f)0.f;

  const unsigned short* arow0 = src + (size_t)(m0 + col) * EDIM;
  const unsigned short* arow1 = src + (size_t)(m0 + 16 + col) * EDIM;
  for (int e0 = 0; e0 < EDIM; e0 += 32) {
    // A fragments (16x32, interleaved-K per ISA table)
    v16bf a0 = ldfrag(arow0 + e0 + sel * 8, arow0 + e0 + 16 + sel * 8);
    v16bf a1 = ldfrag(arow1 + e0 + sel * 8, arow1 + e0 + 16 + sel * 8);
#pragma unroll
    for (int n = 0; n < 8; ++n) {
      // B fragment (32x16): lane col -> W row h = 16n+col; interleaved-K per ISA table
      const unsigned short* wrow = W + (size_t)(n * 16 + col) * EDIM + e0 + sel * 8;
      v16bf bf = ldfrag(wrow, wrow + 16);
      acc[0][n] = WMMA_BF16(a0, bf, acc[0][n]);
      acc[1][n] = WMMA_BF16(a1, bf, acc[1][n]);
    }
  }

  // D layout: VGPR r -> row += r + 8*sel, lane col = output column
#pragma unroll
  for (int h = 0; h < 2; ++h) {
    const int mb = m0 + 16 * h;
    if (which == 2) {
      // store v transposed: vT[b][hd][kv] so stage-2 B-fragments are contiguous
#pragma unroll
      for (int n = 0; n < 8; ++n)
#pragma unroll
        for (int r = 0; r < 8; ++r) {
          int m  = mb + r + 8 * sel;
          int bb = m >> 12;                 // / LKV
          int kv = m & (LKV - 1);
          int hd = n * 16 + col;
          vT[((size_t)bb * HDIM + hd) * LKV + kv] = f2bfu(acc[h][n][r]);
        }
    } else {
      unsigned short* dst = (which == 0) ? qb : kb;
#pragma unroll
      for (int n = 0; n < 8; ++n)
#pragma unroll
        for (int r = 0; r < 8; ++r)
          dst[(size_t)(mb + r + 8 * sel) * HDIM + n * 16 + col] = f2bfu(acc[h][n][r]);
    }
  }
}

// ---------------- Stage 2: flash attention (online softmax, 64 keys/iter) ----
__global__ __launch_bounds__(128) void xattn_flash(
    const unsigned short* __restrict__ qb, const unsigned short* __restrict__ kb,
    const unsigned short* __restrict__ vT, float* __restrict__ out)
{
  // per-wave private P tile (bf16), padded rows -> no cross-wave barrier needed
  __shared__ unsigned short plds[4][16 * PROW];

  const int lane = threadIdx.x & 31;
  const int wave = threadIdx.x >> 5;
  const int qt   = blockIdx.x * 4 + wave;         // 0..1023
  const int b    = qt >> 8;                       // / (LQ/16)
  const int q0   = (qt & 255) * 16;
  const int sel  = lane >> 4;
  const int col  = lane & 15;

  const unsigned short* qbase = qb + (size_t)b * LQ * HDIM;
  const unsigned short* kbase = kb + (size_t)b * LKV * HDIM;
  const unsigned short* vbase = vT + (size_t)b * HDIM * LKV;

  // Q A-fragments over full head dim (sqrt(H) scale already folded into Wq)
  v16bf qa[4];
  {
    const unsigned short* pq = qbase + (size_t)(q0 + col) * HDIM;
#pragma unroll
    for (int j = 0; j < 4; ++j)
      qa[j] = ldfrag(pq + j * 32 + sel * 8, pq + j * 32 + 16 + sel * 8);
  }

  v8f acc[8];
#pragma unroll
  for (int t = 0; t < 8; ++t) acc[t] = (v8f)0.f;
  float M[8], L[8];
#pragma unroll
  for (int r = 0; r < 8; ++r) { M[r] = -__builtin_inff(); L[r] = 0.f; }

  for (int kv0 = 0; kv0 < LKV; kv0 += 64) {
    if (kv0 + 64 < LKV) {                          // CDNA5 global_prefetch_b8
      __builtin_prefetch(kbase + (size_t)(kv0 + 64 + col) * HDIM, 0, 1);
      __builtin_prefetch(vbase + (size_t)col * LKV + kv0 + 64, 0, 1);
    }
    // ---- S = q @ k^T, four 16-key sub-tiles ----
    v8f s[4];
#pragma unroll
    for (int c = 0; c < 4; ++c) s[c] = (v8f)0.f;
#pragma unroll
    for (int j = 0; j < 4; ++j) {
#pragma unroll
      for (int c = 0; c < 4; ++c) {
        const unsigned short* pk =
            kbase + (size_t)(kv0 + c * 16 + col) * HDIM + j * 32 + sel * 8;
        v16bf bf = ldfrag(pk, pk + 16);
        s[c] = WMMA_BF16(qa[j], bf, s[c]);
      }
    }
    // ---- online softmax (rows live across each 16-lane half) ----
    float mnew[8];
#pragma unroll
    for (int r = 0; r < 8; ++r) {
      float m = fmaxf(fmaxf(s[0][r], s[1][r]), fmaxf(s[2][r], s[3][r]));
#pragma unroll
      for (int msk = 1; msk < 16; msk <<= 1)
        m = fmaxf(m, __shfl_xor(m, msk, 32));
      mnew[r] = fmaxf(M[r], m);
    }
    v8f av;                                        // alpha per row (vector -> pk ops)
#pragma unroll
    for (int r = 0; r < 8; ++r) {
      av[r] = __expf(M[r] - mnew[r]);
      M[r]  = mnew[r];
    }
#pragma unroll
    for (int c = 0; c < 4; ++c)
#pragma unroll
      for (int r = 0; r < 8; ++r)
        s[c][r] = __expf(s[c][r] - mnew[r]);
    v8f rsv = (s[0] + s[1]) + (s[2] + s[3]);       // v_pk_add_f32
#pragma unroll
    for (int r = 0; r < 8; ++r) {
      float rs = rsv[r];
#pragma unroll
      for (int msk = 1; msk < 16; msk <<= 1)
        rs += __shfl_xor(rs, msk, 32);
      L[r] = L[r] * av[r] + rs;
    }
#pragma unroll
    for (int t = 0; t < 8; ++t) acc[t] = acc[t] * av;   // v_pk_mul_f32
    // ---- P: C/D layout -> LDS (row-major, padded) -> A-fragment layout ----
    // wave-private LDS slice + in-order LDS pipeline => no s_barrier needed;
    // wave_barrier() only stops compiler reordering (emits no code).
    __builtin_amdgcn_wave_barrier();
#pragma unroll
    for (int r = 0; r < 8; ++r) {
      int row = r + 8 * sel;
#pragma unroll
      for (int c = 0; c < 4; ++c)
        plds[wave][row * PROW + c * 16 + col] = f2bfu(s[c][r]);
    }
    __builtin_amdgcn_wave_barrier();
    const unsigned short* pp = &plds[wave][(lane & 15) * PROW];
    v16bf pa0 = ldfrag(pp + sel * 8,      pp + 16 + sel * 8);
    v16bf pa1 = ldfrag(pp + 32 + sel * 8, pp + 48 + sel * 8);
    __builtin_amdgcn_wave_barrier();
    // ---- O += P @ V over 8 head-dim sub-tiles ----
#pragma unroll
    for (int t = 0; t < 8; ++t) {
      const unsigned short* pv = vbase + (size_t)(t * 16 + col) * LKV + kv0 + sel * 8;
      v16bf vb0 = ldfrag(pv, pv + 16);
      v16bf vb1 = ldfrag(pv + 32, pv + 48);
      acc[t] = WMMA_BF16(pa0, vb0, acc[t]);
      acc[t] = WMMA_BF16(pa1, vb1, acc[t]);
    }
  }

  // ---- normalize and store (f32 out) ----
  v8f inv;
#pragma unroll
  for (int r = 0; r < 8; ++r) inv[r] = 1.0f / L[r];
  float* obase = out + ((size_t)b * LQ + q0) * HDIM;
#pragma unroll
  for (int t = 0; t < 8; ++t) {
    v8f o = acc[t] * inv;                          // v_pk_mul_f32
#pragma unroll
    for (int r = 0; r < 8; ++r) {
      int row = r + 8 * sel;
      obase[(size_t)row * HDIM + t * 16 + col] = o[r];
    }
  }
}

// ---------------- launch ----------------
extern "C" void kernel_launch(void* const* d_in, const int* in_sizes, int n_in,
                              void* d_out, int out_size, void* d_ws, size_t ws_size,
                              hipStream_t stream) {
  const float* x  = (const float*)d_in[0];
  const float* z  = (const float*)d_in[1];
  const float* Wq = (const float*)d_in[2];
  const float* Wk = (const float*)d_in[3];
  const float* Wv = (const float*)d_in[4];

  // workspace layout (bf16): xb | zb | Wb(3) | qb | kb | vT   (~77 MB)
  const size_t nX = (size_t)BATCH * LQ * EDIM;    // 16.7M
  const size_t nW = (size_t)HDIM * EDIM;          // 131072
  const size_t nQ = (size_t)BATCH * LQ * HDIM;    // 2.1M
  unsigned short* xb = (unsigned short*)d_ws;
  unsigned short* zb = xb + nX;
  unsigned short* Wb = zb + nX;                   // [3][H][E]
  unsigned short* qb = Wb + 3 * nW;
  unsigned short* kb = qb + nQ;
  unsigned short* vT = kb + nQ;

  const float sqrtH = 11.313708498984761f;        // reference MULTIPLIES by sqrt(H)

  // Stage 0: convert everything to bf16 (Wq pre-scaled by sqrt(H))
  xattn_cvt_bf16<<<dim3(nX / 2048), dim3(256), 0, stream>>>(x, xb, 1.0f);
  xattn_cvt_bf16<<<dim3(nX / 2048), dim3(256), 0, stream>>>(z, zb, 1.0f);
  xattn_cvt_bf16<<<dim3(nW / 2048), dim3(256), 0, stream>>>(Wq, Wb, sqrtH);
  xattn_cvt_bf16<<<dim3(nW / 2048), dim3(256), 0, stream>>>(Wk, Wb + nW, 1.0f);
  xattn_cvt_bf16<<<dim3(nW / 2048), dim3(256), 0, stream>>>(Wv, Wb + 2 * nW, 1.0f);

  // Stage 1: q, k, vT projections (bf16 GEMM, L2-resident operands)
  dim3 blk(128);
  dim3 g1((BATCH * LQ) / 32 / 4, 3);              // 128 x 3 blocks, 4 waves each
  xattn_qkv_proj<<<g1, blk, 0, stream>>>(xb, zb, Wb, qb, kb, vT);

  // Stage 2: flash attention
  dim3 g2((BATCH * LQ) / 16 / 4);                 // 256 blocks, 4 waves each
  xattn_flash<<<g2, blk, 0, stream>>>(qb, kb, vT, (float*)d_out);
}
